// GroupedQueryAttention_87101936763278
// MI455X (gfx1250) — compile-verified, matrix operand order checked by reading, not run on hardware
//
#include <hip/hip_runtime.h>
#include <hip/hip_bf16.h>

// ---------------------------------------------------------------------------
// GQA attention block for MI455X (gfx1250, wave32, WMMA 16x16x32 bf16 + TDM).
// ---------------------------------------------------------------------------

typedef __attribute__((ext_vector_type(16))) __bf16 v16bf;
typedef __attribute__((ext_vector_type(8)))  __bf16 v8bf;
typedef __attribute__((ext_vector_type(8)))  float  v8f;
typedef __attribute__((ext_vector_type(4)))  unsigned int v4u;
typedef __attribute__((ext_vector_type(8)))  int v8i;
typedef __attribute__((ext_vector_type(4)))  int v4i;

#define WMMA_BF16(a, b, c) \
  __builtin_amdgcn_wmma_f32_16x16x32_bf16(false, (a), false, (b), (short)0, (c), false, false)

__device__ __forceinline__ v8f zero8() {
  v8f z;
#pragma unroll
  for (int i = 0; i < 8; ++i) z[i] = 0.0f;
  return z;
}

// 16x32 A/B fragment for one lane (ISA 16-bit layout): 8 contiguous bf16 at p0
// (K {0..7} of this lane's half) + 8 contiguous bf16 at p1 (K {16..23}).
__device__ __forceinline__ v16bf load_frag(const __bf16* p0, const __bf16* p1) {
  v8bf a = *(const v8bf*)p0;
  v8bf b = *(const v8bf*)p1;
  v16bf r;
#pragma unroll
  for (int i = 0; i < 8; ++i) { r[i] = a[i]; r[i + 8] = b[i]; }
  return r;
}

__device__ __forceinline__ float red_max16(float v) {
#pragma unroll
  for (int m = 1; m < 16; m <<= 1) v = fmaxf(v, __shfl_xor(v, m, 32));
  return v;
}
__device__ __forceinline__ float red_sum16(float v) {
#pragma unroll
  for (int m = 1; m < 16; m <<= 1) v += __shfl_xor(v, m, 32);
  return v;
}

// ---------------------------------------------------------------------------
// Tensor Data Mover: 2D bf16 tile load, global -> LDS (ISA ch.7/8 D# layout).
// Descriptor group0: count=1 | lds_addr | 57-bit global_addr | type=2.
// Group1: data_size=1 (2B), tensor/tile dims, 48-bit dim0 stride.
// ---------------------------------------------------------------------------
__device__ __forceinline__ void tdm_load_bf16_2d(unsigned lds_off, const void* gptr,
                                                 unsigned d0, unsigned d1,
                                                 unsigned long long stride0) {
  unsigned long long ga = (unsigned long long)(size_t)gptr;
  v4u g0;
  g0[0] = 1u;                                                  // count=1, user mode
  g0[1] = lds_off;                                             // lds_addr (bytes)
  g0[2] = (unsigned)ga;                                        // global_addr[31:0]
  g0[3] = (unsigned)((ga >> 32) & 0x01ffffffu) | (2u << 30);   // addr[56:32] | type=2
  v8i g1;
  g1[0] = (int)(1u << 16);                                     // data_size=1 (2 bytes)
  g1[1] = (int)((d0 & 0xffffu) << 16);                         // tensor_dim0[15:0]
  g1[2] = (int)(((d0 >> 16) & 0xffffu) | ((d1 & 0xffffu) << 16));   // dim0 hi | dim1 lo
  g1[3] = (int)(((d1 >> 16) & 0xffffu) | ((d0 & 0xffffu) << 16));   // dim1 hi | tile_dim0
  g1[4] = (int)(d1 & 0xffffu);                                 // tile_dim1; tile_dim2=0
  g1[5] = (int)(unsigned)(stride0 & 0xffffffffu);              // dim0_stride[31:0]
  g1[6] = (int)(unsigned)((stride0 >> 32) & 0xffffu);          // dim0_stride[47:32]
  g1[7] = 0;                                                   // dim1_stride (unused, 2D)
  v4i z4;
#pragma unroll
  for (int i = 0; i < 4; ++i) z4[i] = 0;
#if defined(__clang_major__) && (__clang_major__ >= 23)
  v8i z8;
#pragma unroll
  for (int i = 0; i < 8; ++i) z8[i] = 0;
  __builtin_amdgcn_tensor_load_to_lds(g0, g1, z4, z4, z8, 0);
#else
  __builtin_amdgcn_tensor_load_to_lds(g0, g1, z4, z4, 0);
#endif
}

// ---------------------------------------------------------------------------
// Elementwise conversion / transpose kernels.
// ---------------------------------------------------------------------------
__global__ void cvt_bf16_kernel(const float* __restrict__ in, __bf16* __restrict__ out, size_t n) {
  size_t i = (size_t)blockIdx.x * blockDim.x + threadIdx.x;
  if (i < n) out[i] = (__bf16)in[i];
}

__global__ void cvt_transpose_kernel(const float* __restrict__ W, __bf16* __restrict__ Wt,
                                     int K, int N) {
  size_t i = (size_t)blockIdx.x * blockDim.x + threadIdx.x;
  if (i < (size_t)K * N) {
    int n = (int)(i % N);
    int k = (int)(i / N);
    Wt[(size_t)n * K + k] = (__bf16)W[i];
  }
}

// V (B,S,4,128) bf16 -> Vt (B,4,128,S) bf16
__global__ void transpose_v_kernel(const __bf16* __restrict__ V, __bf16* __restrict__ Vt, int S) {
  size_t i = (size_t)blockIdx.x * blockDim.x + threadIdx.x;
  int d = (int)(i & 127);
  int h = (int)((i >> 7) & 3);
  size_t bs = i >> 9;
  int s = (int)(bs % S);
  int b = (int)(bs / S);
  Vt[(((size_t)b * 4 + h) * 128 + d) * S + s] = V[i];
}

__global__ void rope_kernel(__bf16* __restrict__ t, const float* __restrict__ cosb,
                            const float* __restrict__ sinb, int S, int H) {
  size_t idx = (size_t)blockIdx.x * blockDim.x + threadIdx.x;  // ((b*S+s)*H+h)*64+i
  int i = (int)(idx & 63);
  int h = (int)((idx >> 6) % H);
  size_t sh = idx / (64u * (unsigned)H);  // b*S + s
  int s = (int)(sh % S);
  size_t base = (sh * H + h) * 128 + 2 * i;
  float c  = cosb[(size_t)s * 64 + i];
  float sn = sinb[(size_t)s * 64 + i];
  float xr = (float)t[base];
  float xi = (float)t[base + 1];
  t[base]     = (__bf16)(xr * c - xi * sn);
  t[base + 1] = (__bf16)(xr * sn + xi * c);
}

// ---------------------------------------------------------------------------
// WMMA GEMM:  C(MxN) = A(MxK) * Bt(NxK)^T. Block = 8 waves, tile 128x128.
// Wave owns 32x64 (2 M-tiles x 4 N-tiles = 8 accumulators): 8 WMMA per
// 6 fragment loads per 32-deep K step.
// ---------------------------------------------------------------------------
template <bool F32OUT>
__global__ __launch_bounds__(256) void wmma_gemm_kernel(const __bf16* __restrict__ A,
                                                        const __bf16* __restrict__ Bt,
                                                        void* __restrict__ Cv,
                                                        int M, int N, int K) {
  const int wave = threadIdx.x >> 5;
  const int lane = threadIdx.x & 31;
  const int half = lane >> 4;
  const int l16  = lane & 15;
  const int m0 = blockIdx.y * 128 + (wave >> 1) * 32;
  const int n0 = blockIdx.x * 128 + (wave & 1) * 64;

  const __bf16* arow0 = A + (size_t)(m0 + l16) * K;
  const __bf16* arow1 = arow0 + (size_t)16 * K;
  const __bf16* brow[4];
#pragma unroll
  for (int nt = 0; nt < 4; ++nt) brow[nt] = Bt + (size_t)(n0 + nt * 16 + l16) * K;

  v8f acc[2][4];
#pragma unroll
  for (int i = 0; i < 2; ++i)
#pragma unroll
    for (int nt = 0; nt < 4; ++nt) acc[i][nt] = zero8();

  for (int kk = 0; kk < K; kk += 32) {
    if (kk + 64 < K) {
      __builtin_prefetch(arow0 + kk + 64, 0, 1);
      __builtin_prefetch(arow1 + kk + 64, 0, 1);
    }
    v16bf a0 = load_frag(arow0 + kk + half * 8, arow0 + kk + 16 + half * 8);
    v16bf a1 = load_frag(arow1 + kk + half * 8, arow1 + kk + 16 + half * 8);
#pragma unroll
    for (int nt = 0; nt < 4; ++nt) {
      v16bf bfr = load_frag(brow[nt] + kk + half * 8, brow[nt] + kk + 16 + half * 8);
      acc[0][nt] = WMMA_BF16(a0, bfr, acc[0][nt]);
      acc[1][nt] = WMMA_BF16(a1, bfr, acc[1][nt]);
    }
  }

#pragma unroll
  for (int i = 0; i < 2; ++i)
#pragma unroll
    for (int nt = 0; nt < 4; ++nt)
#pragma unroll
      for (int v = 0; v < 8; ++v) {
        const int row = m0 + i * 16 + v + 8 * half;  // C/D: M = vgpr + 8*(lane>=16)
        const int col = n0 + nt * 16 + l16;          // N = lane%16
        const float val = acc[i][nt][v];
        if constexpr (F32OUT) {
          ((float*)Cv)[(size_t)row * N + col] = val;
        } else {
          ((__bf16*)Cv)[(size_t)row * N + col] = (__bf16)val;
        }
      }
}

// ---------------------------------------------------------------------------
// Flash attention, TDM-staged. grid = (S/128, B*16); 8 waves / block; wave
// owns 16 query rows. All 8 waves of a block share the same (b,h) K/V stream,
// so wave 0 DMAs each 32-key K tile (32x128) and V tile (128x32) into
// double-buffered LDS with tensor_load_to_lds, synchronized by
// s_wait_tensorcnt + block barrier. Online softmax in C/D layout; P reshaped
// D-layout -> A-layout through a per-wave LDS tile.
// ---------------------------------------------------------------------------
__global__ __launch_bounds__(256) void attn_fwd_kernel(const __bf16* __restrict__ Q,
                                                       const __bf16* __restrict__ Km,
                                                       const __bf16* __restrict__ Vt,
                                                       __bf16* __restrict__ O) {
  constexpr int S = 2048, H = 16, KVH = 4, DH = 128;
  constexpr float SCALE = 0.08838834764831845f;  // 1/sqrt(128)

  __shared__ __align__(16) __bf16 ktile[2][32 * DH];  // 32 keys x 128 d
  __shared__ __align__(16) __bf16 vtile[2][DH * 32];  // 128 d x 32 keys
  __shared__ __align__(16) __bf16 pt[8][16][32];      // per-wave P reshape tile

  const int wave = threadIdx.x >> 5;
  const int lane = threadIdx.x & 31;
  const int half = lane >> 4;
  const int l16  = lane & 15;
  const int b   = blockIdx.y >> 4;
  const int h   = blockIdx.y & 15;
  const int kvh = h >> 2;  // N_REP = 4
  const int q0 = blockIdx.x * 128 + wave * 16;
  const int qmax = q0 + 15;
  const int blkKmax = blockIdx.x * 128 + 127;  // block-uniform causal bound

  // Q fragments: 16 rows x 128 d = 4 A-frags of 16x32
  const __bf16* qrow = Q + ((size_t)(b * S + q0 + l16) * H + h) * DH;
  v16bf qf[4];
#pragma unroll
  for (int f = 0; f < 4; ++f)
    qf[f] = load_frag(qrow + f * 32 + half * 8, qrow + f * 32 + 16 + half * 8);

  const __bf16* Kglob = Km + ((size_t)b * S * KVH + kvh) * DH;   // key r: + r*512
  const __bf16* Vglob = Vt + ((size_t)(b * KVH + kvh) * DH) * S; // dim d: + d*S

  v8f o[8];
#pragma unroll
  for (int nt = 0; nt < 8; ++nt) o[nt] = zero8();
  float mrow[8], lrow[8];
#pragma unroll
  for (int v = 0; v < 8; ++v) { mrow[v] = -3.0e38f; lrow[v] = 0.0f; }

  // Prologue: stage tile kv=0 into buffer 0.
  if (wave == 0) {
    tdm_load_bf16_2d((unsigned)(size_t)(void*)&ktile[0][0], Kglob, DH, 32,
                     (unsigned long long)(KVH * DH));
    tdm_load_bf16_2d((unsigned)(size_t)(void*)&vtile[0][0], Vglob, 32, DH,
                     (unsigned long long)S);
  }

  int p = 0;
  for (int kv = 0; kv <= blkKmax; kv += 32, p ^= 1) {
    if (wave == 0) {
      if (kv + 32 <= blkKmax) {
        // Issue next tile into the other buffer, then wait for current tile.
        tdm_load_bf16_2d((unsigned)(size_t)(void*)&ktile[p ^ 1][0],
                         Kglob + (size_t)(kv + 32) * (KVH * DH), DH, 32,
                         (unsigned long long)(KVH * DH));
        tdm_load_bf16_2d((unsigned)(size_t)(void*)&vtile[p ^ 1][0],
                         Vglob + (kv + 32), 32, DH, (unsigned long long)S);
        __builtin_amdgcn_s_wait_tensorcnt(2);
      } else {
        __builtin_amdgcn_s_wait_tensorcnt(0);
      }
    }
    __syncthreads();

    if (kv <= qmax) {  // wave-uniform: EXEC stays all-ones for WMMA
      // ---- S = Q @ K^T for 32 staged keys (two 16x16 score tiles) ----
      v8f s0 = zero8(), s1 = zero8();
      const __bf16* k0 = &ktile[p][(size_t)l16 * DH];
      const __bf16* k1 = &ktile[p][(size_t)(16 + l16) * DH];
#pragma unroll
      for (int f = 0; f < 4; ++f) {
        v16bf kf = load_frag(k0 + f * 32 + half * 8, k0 + f * 32 + 16 + half * 8);
        s0 = WMMA_BF16(qf[f], kf, s0);
      }
#pragma unroll
      for (int f = 0; f < 4; ++f) {
        v16bf kf = load_frag(k1 + f * 32 + half * 8, k1 + f * 32 + 16 + half * 8);
        s1 = WMMA_BF16(qf[f], kf, s1);
      }

      // ---- scale + causal mask + online softmax (row = vgpr + 8*half) ----
#pragma unroll
      for (int v = 0; v < 8; ++v) {
        const int qi = q0 + v + 8 * half;
        const float a0 = s0[v] * SCALE + (((kv + l16) <= qi) ? 0.0f : -1.0e30f);
        const float a1 = s1[v] * SCALE + (((kv + 16 + l16) <= qi) ? 0.0f : -1.0e30f);
        const float mx = red_max16(fmaxf(a0, a1));
        const float newm = fmaxf(mrow[v], mx);
        const float p0 = __expf(a0 - newm);
        const float p1 = __expf(a1 - newm);
        const float rs = red_sum16(p0 + p1);
        const float alpha = __expf(mrow[v] - newm);
        lrow[v] = lrow[v] * alpha + rs;
        mrow[v] = newm;
#pragma unroll
        for (int nt = 0; nt < 8; ++nt) o[nt][v] *= alpha;
        pt[wave][v + 8 * half][l16]      = (__bf16)p0;
        pt[wave][v + 8 * half][l16 + 16] = (__bf16)p1;
      }

      // ---- reload P as 16x32 A-fragment (same-wave DS ops are in-order) ----
      const __bf16* prow = &pt[wave][l16][0];
      v16bf pa = load_frag(prow + half * 8, prow + 16 + half * 8);

      // ---- O += P @ V (8 output tiles across d=128) ----
#pragma unroll
      for (int nt = 0; nt < 8; ++nt) {
        const __bf16* vrow = &vtile[p][(size_t)(nt * 16 + l16) * 32];
        v16bf vf = load_frag(vrow + half * 8, vrow + 16 + half * 8);
        o[nt] = WMMA_BF16(pa, vf, o[nt]);
      }
    }
    __syncthreads();  // all waves done reading buffer p before it is reused
  }

  // ---- normalize and store bf16 output (B,S,H,128) ----
#pragma unroll
  for (int v = 0; v < 8; ++v) {
    const float inv = 1.0f / lrow[v];
    const int row = q0 + v + 8 * half;
    __bf16* orow = O + ((size_t)(b * S + row) * H + h) * DH;
#pragma unroll
    for (int nt = 0; nt < 8; ++nt)
      orow[nt * 16 + l16] = (__bf16)(o[nt][v] * inv);
  }
}

// ---------------------------------------------------------------------------
// Host-side orchestration
// ---------------------------------------------------------------------------
extern "C" void kernel_launch(void* const* d_in, const int* in_sizes, int n_in,
                              void* d_out, int out_size, void* d_ws, size_t ws_size,
                              hipStream_t stream) {
  (void)in_sizes; (void)n_in; (void)out_size; (void)ws_size;
  constexpr int Bc = 2, Sc = 2048, DM = 2048, Hq = 16, KVHc = 4, DHc = 128;
  constexpr int Mrows = Bc * Sc;          // 4096
  constexpr int NKV = KVHc * DHc;         // 512

  const float* x    = (const float*)d_in[0];
  const float* fcos = (const float*)d_in[1];
  const float* fsin = (const float*)d_in[2];
  // d_in[3] = dense mask; causal mask applied analytically instead.
  const float* wq = (const float*)d_in[4];
  const float* wk = (const float*)d_in[5];
  const float* wv = (const float*)d_in[6];
  const float* wo = (const float*)d_in[7];
  float* out = (float*)d_out;

  char* ws = (char*)d_ws;
  size_t off = 0;
  auto carve = [&](size_t elems) {
    void* p = ws + off;
    off += (elems * sizeof(__bf16) + 255) & ~(size_t)255;
    return (__bf16*)p;
  };
  __bf16* Xb  = carve((size_t)Mrows * DM);   // x in bf16
  __bf16* WQt = carve((size_t)DM * DM);      // wq^T
  __bf16* WKt = carve((size_t)NKV * DM);     // wk^T
  __bf16* WVt = carve((size_t)NKV * DM);     // wv^T
  __bf16* WOt = carve((size_t)DM * DM);      // wo^T
  __bf16* Qb  = carve((size_t)Mrows * DM);   // Q (B,S,16,128)
  __bf16* Kb  = carve((size_t)Mrows * NKV);  // K (B,S,4,128)
  __bf16* Vb  = carve((size_t)Mrows * NKV);  // V (B,S,4,128)
  __bf16* Vtb = carve((size_t)Mrows * NKV);  // V^T (B,4,128,S)
  __bf16* Ab  = carve((size_t)Mrows * DM);   // attention output bf16

  const int TB = 256;

  {
    size_t n = (size_t)Mrows * DM;
    cvt_bf16_kernel<<<(unsigned)((n + TB - 1) / TB), TB, 0, stream>>>(x, Xb, n);
  }
  cvt_transpose_kernel<<<(unsigned)(((size_t)DM * DM) / TB), TB, 0, stream>>>(wq, WQt, DM, DM);
  cvt_transpose_kernel<<<(unsigned)(((size_t)DM * NKV) / TB), TB, 0, stream>>>(wk, WKt, DM, NKV);
  cvt_transpose_kernel<<<(unsigned)(((size_t)DM * NKV) / TB), TB, 0, stream>>>(wv, WVt, DM, NKV);
  cvt_transpose_kernel<<<(unsigned)(((size_t)DM * DM) / TB), TB, 0, stream>>>(wo, WOt, DM, DM);

  // QKV projections (bf16 WMMA GEMM, bf16 store); block tile 128x128
  wmma_gemm_kernel<false><<<dim3(DM / 128, Mrows / 128), TB, 0, stream>>>(Xb, WQt, Qb, Mrows, DM, DM);
  wmma_gemm_kernel<false><<<dim3(NKV / 128, Mrows / 128), TB, 0, stream>>>(Xb, WKt, Kb, Mrows, NKV, DM);
  wmma_gemm_kernel<false><<<dim3(NKV / 128, Mrows / 128), TB, 0, stream>>>(Xb, WVt, Vb, Mrows, NKV, DM);

  rope_kernel<<<(unsigned)(((size_t)Mrows * Hq * 64) / TB), TB, 0, stream>>>(Qb, fcos, fsin, Sc, Hq);
  rope_kernel<<<(unsigned)(((size_t)Mrows * KVHc * 64) / TB), TB, 0, stream>>>(Kb, fcos, fsin, Sc, KVHc);

  transpose_v_kernel<<<(unsigned)(((size_t)Mrows * NKV) / TB), TB, 0, stream>>>(Vb, Vtb, Sc);

  attn_fwd_kernel<<<dim3(Sc / 128, Bc * Hq), TB, 0, stream>>>(Qb, Kb, Vtb, Ab);

  wmma_gemm_kernel<true><<<dim3(DM / 128, Mrows / 128), TB, 0, stream>>>(Ab, WOt, out, Mrows, DM, DM);
}
